// FuncGNN_86191403696814
// MI455X (gfx1250) — matrix operand order repaired and checked by reading, not compiled, not run on hardware
//
#include <hip/hip_runtime.h>
#include <hip/hip_bf16.h>

typedef __attribute__((ext_vector_type(16))) _Float16 v16h;
typedef __attribute__((ext_vector_type(8)))  _Float16 v8h;
typedef __attribute__((ext_vector_type(4)))  _Float16 v4h;
typedef __attribute__((ext_vector_type(2)))  _Float16 v2h;
typedef __attribute__((ext_vector_type(8)))  float    v8f;

#define HID   128
#define NND   50000
#define NEDGE 500000
#define NB    32
#define NTASK 256     // B * TASKS_PER
#define TED   64
#define NCLS  3

// ---------------------------------------------------------------------------
// A-fragment (16x32 f16) from LDS tile sX[64 rows][32 k] (row-major).
// Layout: lanes 0-15 row M=lane hold K{0..15}; lanes 16-31 row
// M=lane-16 hold K{16..31}.  Two contiguous 16B chunks -> ds_load_b128.
__device__ __forceinline__ v16h frag_a(const _Float16* sX, int row, int lane) {
    const _Float16* p = sX + row * 32 + ((lane < 16) ? 0 : 16);
    v16h a;
    ((v8h*)&a)[0] = *(const v8h*)(p);
    ((v8h*)&a)[1] = *(const v8h*)(p + 8);
    return a;
}

// B-fragment (32x16 f16) from TRANSPOSED LDS tile sWt[128 cols][32 k].
// lanes 0-15: col N=lane, K=0..15; lanes 16-31: col N=lane-16, K=16..31.
// 16 contiguous halfs, 32B aligned -> 2x ds_load_b128.
__device__ __forceinline__ v16h frag_b_t(const _Float16* sWt, int ct, int lane) {
    int col = ct * 16 + (lane & 15);
    int kb  = (lane < 16) ? 0 : 16;
    return *(const v16h*)(sWt + col * 32 + kb);
}

// Stage 32xHID chunk of W (rows k0..k0+31) transposed into sWt[col*32 + k].
// thread t owns column t: global reads coalesced across block, v4h LDS stores.
__device__ __forceinline__ void stage_w_t(const float* __restrict__ W, int k0,
                                          _Float16* sWt, int tid) {
    const float* wp = W + (size_t)k0 * HID + tid;
#pragma unroll
    for (int r = 0; r < 32; r += 4) {
        v4h v;
        v[0] = (_Float16)wp[(size_t)(r + 0) * HID];
        v[1] = (_Float16)wp[(size_t)(r + 1) * HID];
        v[2] = (_Float16)wp[(size_t)(r + 2) * HID];
        v[3] = (_Float16)wp[(size_t)(r + 3) * HID];
        *(v4h*)(sWt + tid * 32 + r) = v;
    }
}

// Stage 64x32 chunk of X (fp32 -> f16), packed v2h stores.
__device__ __forceinline__ void stage_x(const float* __restrict__ X, int M, int K,
                                        int rb, int k0, _Float16* sX, int tid) {
#pragma unroll
    for (int i = tid; i < 64 * 16; i += 128) {
        int r = i >> 4, c2 = (i & 15) * 2;
        int gr = rb + r;
        v2h v;
        if (gr < M) {
            const float* p = X + (size_t)gr * K + k0 + c2;
            v[0] = (_Float16)p[0];
            v[1] = (_Float16)p[1];
        } else {
            v[0] = (_Float16)0.f; v[1] = (_Float16)0.f;
        }
        *(v2h*)(sX + r * 32 + c2) = v;
    }
}

// ---------------------------------------------------------------------------
// Generic  out[M x 128] = act(X[M x K] @ W[K x 128] + bias)
// block = 128 threads (4 waves), block does 64 rows, each wave a 16x128 stripe.
__global__ __launch_bounds__(128) void gemm_wmma(
        const float* __restrict__ X, const float* __restrict__ W,
        const float* __restrict__ bias, float* __restrict__ out,
        int M, int K, int do_relu) {
    __shared__ _Float16 sX [64 * 32];
    __shared__ _Float16 sWt[HID * 32];
    const int tid  = threadIdx.x;
    const int wave = tid >> 5, lane = tid & 31;
    const int rb   = blockIdx.x * 64;

    v8f acc[8] = {};
    for (int k0 = 0; k0 < K; k0 += 32) {
        __syncthreads();
        stage_x(X, M, K, rb, k0, sX, tid);
        stage_w_t(W, k0, sWt, tid);
        __syncthreads();
        v16h a = frag_a(sX, wave * 16 + (lane & 15), lane);
#pragma unroll
        for (int ct = 0; ct < 8; ++ct) {
            v16h b = frag_b_t(sWt, ct, lane);
            acc[ct] = __builtin_amdgcn_wmma_f32_16x16x32_f16(
                false, a, false, b, (short)0, acc[ct], false, false);
        }
    }
    // C layout: VGPR v -> row v (lanes 0-15) or v+8 (lanes 16-31), col = lane&15
    const int rbase = rb + wave * 16 + ((lane < 16) ? 0 : 8);
    const int cl = lane & 15;
#pragma unroll
    for (int ct = 0; ct < 8; ++ct) {
        int col = ct * 16 + cl;
        float bv = bias ? bias[col] : 0.0f;
#pragma unroll
        for (int v = 0; v < 8; ++v) {
            int row = rbase + v;
            if (row < M) {
                float val = acc[ct][v] + bv;
                if (do_relu) val = fmaxf(val, 0.0f);
                out[(size_t)row * HID + col] = val;
            }
        }
    }
}

// ---------------------------------------------------------------------------
// Fused: out = emb @ resW + relu(emb @ updW[0:128] + aggr @ updW[128:256] + ub)
__global__ __launch_bounds__(128) void fused_update(
        const float* __restrict__ emb, const float* __restrict__ aggr,
        const float* __restrict__ resW, const float* __restrict__ updW,
        const float* __restrict__ ub, float* __restrict__ out, int M) {
    __shared__ _Float16 sE [64 * 32];
    __shared__ _Float16 sA [64 * 32];
    __shared__ _Float16 sRt[HID * 32];
    __shared__ _Float16 sUt[HID * 32];
    __shared__ _Float16 sVt[HID * 32];
    const int tid  = threadIdx.x;
    const int wave = tid >> 5, lane = tid & 31;
    const int rb   = blockIdx.x * 64;

    v8f uacc[8] = {};
    v8f racc[8] = {};
    for (int k0 = 0; k0 < HID; k0 += 32) {
        __syncthreads();
        stage_x(emb,  M, HID, rb, k0, sE, tid);
        stage_x(aggr, M, HID, rb, k0, sA, tid);
        stage_w_t(resW,             k0, sRt, tid);
        stage_w_t(updW,             k0, sUt, tid);
        stage_w_t(updW + 128 * HID, k0, sVt, tid);
        __syncthreads();
        v16h ae = frag_a(sE, wave * 16 + (lane & 15), lane);
        v16h aa = frag_a(sA, wave * 16 + (lane & 15), lane);
#pragma unroll
        for (int ct = 0; ct < 8; ++ct) {
            v16h bu = frag_b_t(sUt, ct, lane);
            uacc[ct] = __builtin_amdgcn_wmma_f32_16x16x32_f16(
                false, ae, false, bu, (short)0, uacc[ct], false, false);
            v16h bv = frag_b_t(sVt, ct, lane);
            uacc[ct] = __builtin_amdgcn_wmma_f32_16x16x32_f16(
                false, aa, false, bv, (short)0, uacc[ct], false, false);
            v16h br = frag_b_t(sRt, ct, lane);
            racc[ct] = __builtin_amdgcn_wmma_f32_16x16x32_f16(
                false, ae, false, br, (short)0, racc[ct], false, false);
        }
    }
    const int rbase = rb + wave * 16 + ((lane < 16) ? 0 : 8);
    const int cl = lane & 15;
#pragma unroll
    for (int ct = 0; ct < 8; ++ct) {
        int col = ct * 16 + cl;
        float bv = ub[col];
#pragma unroll
        for (int v = 0; v < 8; ++v) {
            int row = rbase + v;
            if (row < M)
                out[(size_t)row * HID + col] =
                    racc[ct][v] + fmaxf(uacc[ct][v] + bv, 0.0f);
        }
    }
}

// ---------------------------------------------------------------------------
__global__ void zero_f(float* p, int n) {
    int i = blockIdx.x * blockDim.x + threadIdx.x;
    if (i < n) p[i] = 0.0f;
}

__global__ void edge_dist(const int* __restrict__ ei, const float* __restrict__ x,
                          float* __restrict__ dist) {
    int e = blockIdx.x * blockDim.x + threadIdx.x;
    if (e >= NEDGE) return;
    int r = ei[e], c = ei[NEDGE + e];
    float dx = x[r * 3 + 0] - x[c * 3 + 0];
    float dy = x[r * 3 + 1] - x[c * 3 + 1];
    float dz = x[r * 3 + 2] - x[c * 3 + 2];
    dist[e] = dx * dx + dy * dy + dz * dz;
}

// msg = relu(A[row] + Bm[col] + dist * w_last + mb) ; aggr[col] += msg
// 32 lanes per edge, 4 floats per lane (float4 path).
__global__ void edge_msg_aggr(const int* __restrict__ ei, const float* __restrict__ dist,
                              const float4* __restrict__ A, const float4* __restrict__ Bm,
                              const float* __restrict__ wlast, const float* __restrict__ mb,
                              float* __restrict__ aggr) {
    int idx = blockIdx.x * blockDim.x + threadIdx.x;
    int e = idx >> 5;
    if (e >= NEDGE) return;
    int j = idx & 31;
    int r = ei[e], c = ei[NEDGE + e];
    float d = dist[e];
    float4 a = A[(size_t)r * 32 + j];
    float4 b = Bm[(size_t)c * 32 + j];
    float4 w = ((const float4*)wlast)[j];
    float4 bb = ((const float4*)mb)[j];
    float v0 = fmaxf(a.x + b.x + d * w.x + bb.x, 0.0f);
    float v1 = fmaxf(a.y + b.y + d * w.y + bb.y, 0.0f);
    float v2 = fmaxf(a.z + b.z + d * w.z + bb.z, 0.0f);
    float v3 = fmaxf(a.w + b.w + d * w.w + bb.w, 0.0f);
    float* dst = aggr + (size_t)c * HID + j * 4;
    atomicAdd(dst + 0, v0);
    atomicAdd(dst + 1, v1);
    atomicAdd(dst + 2, v2);
    atomicAdd(dst + 3, v3);
}

__global__ void pool_accum(const float* __restrict__ emb, const int* __restrict__ batch,
                           float* __restrict__ pooled, float* __restrict__ counts) {
    int idx = blockIdx.x * blockDim.x + threadIdx.x;
    int node = idx >> 5;
    if (node >= NND) return;
    int j = idx & 31;
    int b = batch[node];
    float4 v = ((const float4*)emb)[(size_t)node * 32 + j];
    float* dst = pooled + (size_t)b * HID + j * 4;
    atomicAdd(dst + 0, v.x);
    atomicAdd(dst + 1, v.y);
    atomicAdd(dst + 2, v.z);
    atomicAdd(dst + 3, v.w);
    if (j == 0) atomicAdd(&counts[b], 1.0f);
}

// feat[i] = concat(pooled[proto]/count, task_emb[tid]) -> 256 x 192
__global__ void build_feat(const float* __restrict__ pooled, const float* __restrict__ counts,
                           const float* __restrict__ task_emb, const int* __restrict__ tasks,
                           float* __restrict__ feat) {
    int idx = blockIdx.x * blockDim.x + threadIdx.x;
    if (idx >= NTASK * (HID + TED)) return;
    int i = idx / (HID + TED), c = idx % (HID + TED);
    int proto = tasks[i * 2], tsk = tasks[i * 2 + 1];
    float v;
    if (c < HID) v = pooled[proto * HID + c] / fmaxf(counts[proto], 1.0f);
    else         v = task_emb[(size_t)tsk * TED + (c - HID)];
    feat[idx] = v;
}

__global__ void final_mlp(const float* __restrict__ z, const float* __restrict__ W3,
                          const float* __restrict__ b3, float* __restrict__ out) {
    int idx = blockIdx.x * blockDim.x + threadIdx.x;
    if (idx >= NTASK * NCLS) return;
    int i = idx / NCLS, c = idx % NCLS;
    float s = b3[c];
#pragma unroll 8
    for (int k = 0; k < HID; ++k) s += z[i * HID + k] * W3[k * NCLS + c];
    out[idx] = s;
}

// ---------------------------------------------------------------------------
extern "C" void kernel_launch(void* const* d_in, const int* in_sizes, int n_in,
                              void* d_out, int out_size, void* d_ws, size_t ws_size,
                              hipStream_t stream) {
    const float* h        = (const float*)d_in[0];
    const float* x        = (const float*)d_in[1];
    const int*   ei       = (const int*)  d_in[2];
    /* d_in[3] edge_attr unused by reference */
    const int*   batch    = (const int*)  d_in[4];
    const int*   tasks    = (const int*)  d_in[5];
    /* d_in[6] batch_size scalar == NB */
    const float* embed_W  = (const float*)d_in[7];
    const float* embed_b  = (const float*)d_in[8];
    const float* res_W    = (const float*)d_in[9];
    const float* msg_W    = (const float*)d_in[10];
    const float* msg_b    = (const float*)d_in[11];
    const float* upd_W    = (const float*)d_in[12];
    const float* upd_b    = (const float*)d_in[13];
    const float* W0 = (const float*)d_in[14]; const float* b0 = (const float*)d_in[15];
    const float* W1 = (const float*)d_in[16]; const float* b1 = (const float*)d_in[17];
    const float* W2 = (const float*)d_in[18]; const float* b2 = (const float*)d_in[19];
    const float* W3 = (const float*)d_in[20]; const float* b3 = (const float*)d_in[21];
    const float* task_emb = (const float*)d_in[22];
    float* out = (float*)d_out;

    // workspace carve (floats)
    float* ws = (float*)d_ws;
    size_t off = 0;
    float* emb0   = ws + off; off += (size_t)NND * HID;
    float* emb1   = ws + off; off += (size_t)NND * HID;
    float* Abuf   = ws + off; off += (size_t)NND * HID;
    float* Bbuf   = ws + off; off += (size_t)NND * HID;
    float* aggr   = ws + off; off += (size_t)NND * HID;
    float* dist   = ws + off; off += (size_t)NEDGE;
    float* pooled = ws + off; off += (size_t)NB * HID;
    float* counts = ws + off; off += (size_t)NB;
    float* feat   = ws + off; off += (size_t)NTASK * (HID + TED);
    float* z0     = ws + off; off += (size_t)NTASK * HID;
    float* z1     = ws + off; off += (size_t)NTASK * HID;

    const int TPB = 256;
    dim3 gGemmN((NND + 63) / 64), b128(128);

    // 1) edge squared distances
    edge_dist<<<(NEDGE + TPB - 1) / TPB, TPB, 0, stream>>>(ei, x, dist);

    // 2) emb = h @ embed_W + embed_b        (K = 64)
    gemm_wmma<<<gGemmN, b128, 0, stream>>>(h, embed_W, embed_b, emb0, NND, 64, 0);

    float* cur = emb0;
    float* nxt = emb1;
    for (int l = 0; l < 2; ++l) {
        const float* mW    = msg_W + (size_t)l * 257 * HID;
        const float* wlast = mW + (size_t)256 * HID;
        const float* mb    = msg_b + (size_t)l * HID;

        // split concat-GEMM: A = emb @ mW[0:128], Bm = emb @ mW[128:256]
        gemm_wmma<<<gGemmN, b128, 0, stream>>>(cur, mW,              nullptr, Abuf, NND, HID, 0);
        gemm_wmma<<<gGemmN, b128, 0, stream>>>(cur, mW + 128 * HID,  nullptr, Bbuf, NND, HID, 0);

        zero_f<<<(NND * HID + TPB - 1) / TPB, TPB, 0, stream>>>(aggr, NND * HID);
        edge_msg_aggr<<<(NEDGE * 32) / TPB, TPB, 0, stream>>>(
            ei, dist, (const float4*)Abuf, (const float4*)Bbuf, wlast, mb, aggr);

        // emb' = emb @ res_W[l] + relu([emb,aggr] @ upd_W[l] + upd_b[l])
        fused_update<<<gGemmN, b128, 0, stream>>>(
            cur, aggr, res_W + (size_t)l * HID * HID, upd_W + (size_t)l * 256 * HID,
            upd_b + (size_t)l * HID, nxt, NND);
        float* t = cur; cur = nxt; nxt = t;
    }

    // mean pool per graph
    zero_f<<<(NB * HID + NB + TPB - 1) / TPB, TPB, 0, stream>>>(pooled, NB * HID + NB);
    pool_accum<<<(NND * 32 + TPB - 1) / TPB, TPB, 0, stream>>>(cur, batch, pooled, counts);

    // feat = concat(pooled[proto], task_emb[tid])
    build_feat<<<(NTASK * (HID + TED) + TPB - 1) / TPB, TPB, 0, stream>>>(
        pooled, counts, task_emb, tasks, feat);

    // task MLP
    dim3 gT((NTASK + 63) / 64);
    gemm_wmma<<<gT, b128, 0, stream>>>(feat, W0, b0, z0, NTASK, HID + TED, 1);
    gemm_wmma<<<gT, b128, 0, stream>>>(z0,   W1, b1, z1, NTASK, HID, 1);
    gemm_wmma<<<gT, b128, 0, stream>>>(z1,   W2, b2, z0, NTASK, HID, 1);
    final_mlp<<<(NTASK * NCLS + TPB - 1) / TPB, TPB, 0, stream>>>(z0, W3, b3, out);
}
